// DecoderLayer_73263552135573
// MI455X (gfx1250) — matrix operand order repaired and checked by reading, not compiled, not run on hardware
//
#include <hip/hip_runtime.h>
#include <hip/hip_bf16.h>

// ---------------------------------------------------------------------------
// Dimensions (compile-time constants from the reference)
// ---------------------------------------------------------------------------
#define BB   8
#define II   48
#define JJ   512
#define KK   16
#define DD   512
#define HH   8
#define DFF  2048
#define DHID 256          // DH
#define DK   32           // cross-attn head dim
#define DKS  64           // self-attn head dim
#define NROW (BB*II)      // 384

typedef __attribute__((ext_vector_type(16))) _Float16 v16h;
typedef __attribute__((ext_vector_type(8)))  float    v8f;
typedef __attribute__((ext_vector_type(4)))  int      vi4;

#define AS1 __attribute__((address_space(1)))
#define AS3 __attribute__((address_space(3)))

#if __has_builtin(__builtin_amdgcn_global_load_async_to_lds_b128)
#define HAVE_ASYNC_LDS 1
#else
#define HAVE_ASYNC_LDS 0
#endif

// ---------------------------------------------------------------------------
// Pack a row-major f32 weight [K][N] into the WMMA B-operand f16 layout:
// tile = kt*Ntiles + nt ; within tile: lane l holds column n = nt*16 + (l&15),
// 16 halves e -> k = kt*32 + (e>>3)*16 + (l>>4)*8 + (e&7).  (CDNA5 ISA §7.12.2)
// ---------------------------------------------------------------------------
__global__ void pack_b_kernel(const float* __restrict__ W, _Float16* __restrict__ out,
                              int Kd, int N) {
  int Ntiles = N >> 4;
  size_t total = (size_t)Kd * N;
  for (size_t idx = (size_t)blockIdx.x * blockDim.x + threadIdx.x; idx < total;
       idx += (size_t)gridDim.x * blockDim.x) {
    int e    = (int)(idx & 15);
    int lane = (int)((idx >> 4) & 31);
    size_t tile = idx >> 9;                 // 32 lanes * 16 halves = 512 / tile
    int nt = (int)(tile % Ntiles);
    int kt = (int)(tile / Ntiles);
    int k = kt * 32 + ((e >> 3) << 4) + ((lane >> 4) << 3) + (e & 7);
    int n = nt * 16 + (lane & 15);
    out[idx] = (_Float16)W[(size_t)k * N + n];
  }
}

// ---------------------------------------------------------------------------
// Generic WMMA GEMM: C[M,N] = act(A[M,K] @ B[K,N] + bias) + res
// A: row-major f32 (converted to f16 in-register); B: pre-packed f16.
// Block = 128 threads = 4 waves; each wave computes one 16x16 tile.
// Grid = (M/16, N/64).
// ---------------------------------------------------------------------------
template<int ACT>   // 0=none, 1=relu, 2=gelu(tanh)
__global__ __launch_bounds__(128)
void gemm_wmma_kernel(const float* __restrict__ A, const _Float16* __restrict__ Bp,
                      const float* __restrict__ bias, const float* __restrict__ res,
                      float* __restrict__ C, int M, int Kd, int N) {
  const int lane = threadIdx.x & 31;
  const int wave = threadIdx.x >> 5;
  const int mt = blockIdx.x;
  const int nt = blockIdx.y * 4 + wave;
  const int Ntiles = N >> 4;

  const int m0 = mt * 16 + (lane & 15);       // A row for this lane
  const int kh = (lane >> 4) << 3;            // 0 or 8: K-half select (A layout)
  v8f acc = {0.f,0.f,0.f,0.f,0.f,0.f,0.f,0.f};

  const int Ksteps = Kd >> 5;
  for (int ks = 0; ks < Ksteps; ++ks) {
    const float* ap = A + (size_t)m0 * Kd + ks * 32 + kh;
    v16h af;
#pragma unroll
    for (int e = 0; e < 8; ++e) af[e] = (_Float16)ap[e];
#pragma unroll
    for (int e = 0; e < 8; ++e) af[8 + e] = (_Float16)ap[16 + e];
    v16h bf = *(const v16h*)(Bp + (((size_t)ks * Ntiles + nt) * 32 + lane) * 16);
    acc = __builtin_amdgcn_wmma_f32_16x16x32_f16(false, af, false, bf,
                                                 (short)0, acc, false, false);
  }

  const int n = nt * 16 + (lane & 15);
  const float bv = bias ? bias[n] : 0.f;
#pragma unroll
  for (int r = 0; r < 8; ++r) {
    int m = mt * 16 + r + ((lane >> 4) << 3);   // C layout: M = r + 8*(lane/16)
    float v = acc[r] + bv;
    if (ACT == 1) v = fmaxf(v, 0.f);
    else if (ACT == 2) {
      float t = 0.7978845608028654f * (v + 0.044715f * v * v * v);
      v = 0.5f * v * (1.f + tanhf(t));
    }
    if (res) v += res[(size_t)m * N + n];
    C[(size_t)m * N + n] = v;
  }
}

// ---------------------------------------------------------------------------
// LayerNorm (+ optional sinusoidal PE) ; one block per row of length 512
// ---------------------------------------------------------------------------
__global__ __launch_bounds__(256)
void ln_kernel(const float* __restrict__ x, const float* __restrict__ g,
               const float* __restrict__ b, float* __restrict__ y, int addpe) {
  const int row = blockIdx.x;
  const float* xr = x + (size_t)row * DD;
  __shared__ float sred[256];
  float s = 0.f;
  for (int c = threadIdx.x; c < DD; c += 256) s += xr[c];
  sred[threadIdx.x] = s; __syncthreads();
  for (int off = 128; off > 0; off >>= 1) {
    if (threadIdx.x < off) sred[threadIdx.x] += sred[threadIdx.x + off];
    __syncthreads();
  }
  float mean = sred[0] / (float)DD;
  __syncthreads();
  float s2 = 0.f;
  for (int c = threadIdx.x; c < DD; c += 256) { float d = xr[c] - mean; s2 += d * d; }
  sred[threadIdx.x] = s2; __syncthreads();
  for (int off = 128; off > 0; off >>= 1) {
    if (threadIdx.x < off) sred[threadIdx.x] += sred[threadIdx.x + off];
    __syncthreads();
  }
  float rstd = rsqrtf(sred[0] / (float)DD + 1e-5f);
  for (int c = threadIdx.x; c < DD; c += 256) {
    float v = (xr[c] - mean) * rstd * g[c] + b[c];
    if (addpe) {
      int pos = row % II;
      int half = c >> 1;
      float ang = (float)pos * __powf(10000.f, -(2.f * (float)half) / (float)DD);
      v += (c & 1) ? __cosf(ang) : __sinf(ang);
    }
    y[(size_t)row * DD + c] = v;
  }
}

// ---------------------------------------------------------------------------
// Self-attention core: per (b,h) block.  qkv row layout: [H][3*DKS] per token.
// ---------------------------------------------------------------------------
__global__ __launch_bounds__(256)
void self_attn_kernel(const float* __restrict__ qkv, const int* __restrict__ mask,
                      float* __restrict__ out) {
  const int b = blockIdx.x >> 3, h = blockIdx.x & 7;
  __shared__ float s[II * II];
  const float* base = qkv + (size_t)b * II * (3 * DD);
  for (int p = threadIdx.x; p < II * II; p += 256) {
    int i1 = p / II, i2 = p % II;
    const float* q = base + (size_t)i1 * (3 * DD) + h * (3 * DKS);
    const float* k = base + (size_t)i2 * (3 * DD) + h * (3 * DKS) + DKS;
    float a = 0.f;
    for (int d = 0; d < DKS; ++d) a += q[d] * k[d];
    a *= 0.125f;                               // DKS^-0.5
    if (mask[i1 * II + i2] == 0) a = -1e9f;
    s[p] = a;
  }
  __syncthreads();
  if (threadIdx.x < II) {
    int i1 = threadIdx.x;
    float mx = -1e30f;
    for (int i2 = 0; i2 < II; ++i2) mx = fmaxf(mx, s[i1 * II + i2]);
    float sum = 0.f;
    for (int i2 = 0; i2 < II; ++i2) { float e = __expf(s[i1 * II + i2] - mx); s[i1 * II + i2] = e; sum += e; }
    float inv = 1.f / sum;
    for (int i2 = 0; i2 < II; ++i2) s[i1 * II + i2] *= inv;
  }
  __syncthreads();
  for (int p = threadIdx.x; p < II * DKS; p += 256) {
    int i1 = p / DKS, d = p % DKS;
    float a = 0.f;
    for (int i2 = 0; i2 < II; ++i2)
      a += s[i1 * II + i2] * base[(size_t)i2 * (3 * DD) + h * (3 * DKS) + 2 * DKS + d];
    out[((size_t)b * II + i1) * DD + h * DKS + d] = a;
  }
}

// ---------------------------------------------------------------------------
// Local cross-attn: per (b,i) block.  16 gathered neighbors, H=8, DK=32.
// Writes concat row [agg(256) | distance(512)] of width 768.
// ---------------------------------------------------------------------------
__global__ __launch_bounds__(256)
void local_attn_kernel(const float* __restrict__ q, const float* __restrict__ kv,
                       const float* __restrict__ p, const int* __restrict__ idx,
                       const float* __restrict__ dist, float* __restrict__ concat) {
  const int bi = blockIdx.x;
  const int b = bi / II;
  __shared__ float sq[DHID];
  __shared__ float slog[KK * HH];
  __shared__ float sw[KK * HH];
  sq[threadIdx.x] = q[(size_t)bi * DHID + threadIdx.x];
  __syncthreads();
  if (threadIdx.x < KK * HH) {
    int jl = threadIdx.x & 15, h = threadIdx.x >> 4;
    int id = idx[bi * KK + jl];
    const float* kr = kv + ((size_t)b * JJ + id) * (2 * DHID) + h * DK;
    const float* pr = p + ((size_t)bi * KK + jl) * (2 * DHID) + h * (2 * DK);
    float a = 0.f;
    for (int d = 0; d < DK; ++d) a += sq[h * DK + d] * (kr[d] + pr[d]);
    slog[jl * HH + h] = a * 0.17677669529663687f;   // DK^-0.5
  }
  __syncthreads();
  if (threadIdx.x < HH) {
    int h = threadIdx.x;
    float mx = -1e30f;
    for (int jl = 0; jl < KK; ++jl) mx = fmaxf(mx, slog[jl * HH + h]);
    float sum = 0.f;
    for (int jl = 0; jl < KK; ++jl) { float e = __expf(slog[jl * HH + h] - mx); sw[jl * HH + h] = e; sum += e; }
    float inv = 1.f / sum;
    for (int jl = 0; jl < KK; ++jl) sw[jl * HH + h] *= inv;
  }
  __syncthreads();
  {
    int h = threadIdx.x >> 5, d = threadIdx.x & 31;
    float a = 0.f;
    for (int jl = 0; jl < KK; ++jl) {
      int id = idx[bi * KK + jl];
      a += sw[jl * HH + h] *
           (kv[((size_t)b * JJ + id) * (2 * DHID) + DHID + h * DK + d] +
            p[((size_t)bi * KK + jl) * (2 * DHID) + h * (2 * DK) + DK + d]);
    }
    concat[(size_t)bi * (DHID + DD) + h * DK + d] = a;
  }
  for (int c = threadIdx.x; c < DD; c += 256)
    concat[(size_t)bi * (DHID + DD) + DHID + c] = dist[(size_t)bi * DD + c];
}

// ---------------------------------------------------------------------------
// Fused global cross-attn: per (b,i) block, 8 waves.  Streams global_dist
// j-tiles (16 rows x 256 f32, fully contiguous = 16KB) into LDS once
// (async-to-LDS when available), then:
//   GEMM1 (WMMA): hidden = relu(tile @ W1 + b1)    -> LDS (f16)
//   GEMM2 (WMMA): p      = hidden @ W2 + b2        -> LDS (f32)
//   logits / online softmax / V accumulation (VALU)
// ---------------------------------------------------------------------------
__global__ __launch_bounds__(256)
void global_attn_kernel(const float* __restrict__ q, const float* __restrict__ kv,
                        const float* __restrict__ gd,
                        const _Float16* __restrict__ W1p, const float* __restrict__ b1,
                        const _Float16* __restrict__ W2p, const float* __restrict__ b2,
                        const unsigned char* __restrict__ mask, float* __restrict__ agg) {
  const int bi = blockIdx.x;
  const int b = bi / II;
  const int lane = threadIdx.x & 31;
  const int wave = threadIdx.x >> 5;
  const int t = threadIdx.x;

  __shared__ float    s_a32[16 * DHID];      // 16 KB staged A tile (f32)
  __shared__ _Float16 s_hid[16 * DHID];      // 8 KB
  __shared__ float    s_p[16 * (2 * DHID)];  // 32 KB
  __shared__ float    s_q[DHID];
  __shared__ float    s_log[16 * HH];
  __shared__ float    s_w[16 * HH];
  __shared__ float    s_m[HH], s_s[HH], s_sc[HH];
  __shared__ float    s_acc[HH * DK];

  s_q[t] = q[(size_t)bi * DHID + t];
  s_acc[t] = 0.f;
  if (t < HH) { s_m[t] = -1e30f; s_s[t] = 0.f; }
  __syncthreads();

  const float* gdb = gd + (size_t)bi * JJ * DHID;

  for (int jt = 0; jt < JJ / 16; ++jt) {
    // ---- stage A tile (16x256 f32, contiguous) into LDS ----
    {
      const float* tile = gdb + (size_t)jt * 16 * DHID;
#if HAVE_ASYNC_LDS
#pragma unroll
      for (int it = 0; it < 4; ++it) {
        int c = t + it * 256;                      // 1024 x b128 chunks
        AS1 vi4* gsrc = (AS1 vi4*)(uintptr_t)(tile + c * 4);
        AS3 vi4* ldst = (AS3 vi4*)(unsigned int)(uintptr_t)(s_a32 + c * 4);
        __builtin_amdgcn_global_load_async_to_lds_b128(gsrc, ldst, 0, 0);
      }
#if __has_builtin(__builtin_amdgcn_s_wait_asynccnt)
      __builtin_amdgcn_s_wait_asynccnt(0);
#else
      asm volatile("s_wait_asynccnt 0" ::: "memory");
#endif
#else
      for (int c = t; c < 16 * DHID; c += 256) s_a32[c] = tile[c];
#endif
    }
    __syncthreads();

    // ---- GEMM1: 16 N-tiles of 16; each wave does 2. K = 256 -> 8 steps ----
    {
      const int m0 = lane & 15;
      const int khalf = (lane >> 4) << 3;
      const int nt0 = wave * 2;
      v8f acc0 = {0.f,0.f,0.f,0.f,0.f,0.f,0.f,0.f};
      v8f acc1 = {0.f,0.f,0.f,0.f,0.f,0.f,0.f,0.f};
      for (int ks = 0; ks < DHID / 32; ++ks) {
        const float* ap = s_a32 + m0 * DHID + ks * 32 + khalf;
        v16h af;
#pragma unroll
        for (int e = 0; e < 8; ++e) af[e] = (_Float16)ap[e];
#pragma unroll
        for (int e = 0; e < 8; ++e) af[8 + e] = (_Float16)ap[16 + e];
        v16h bf0 = *(const v16h*)(W1p + (((size_t)ks * (DHID / 16) + nt0) * 32 + lane) * 16);
        v16h bf1 = *(const v16h*)(W1p + (((size_t)ks * (DHID / 16) + nt0 + 1) * 32 + lane) * 16);
        acc0 = __builtin_amdgcn_wmma_f32_16x16x32_f16(false, af, false, bf0,
                                                      (short)0, acc0, false, false);
        acc1 = __builtin_amdgcn_wmma_f32_16x16x32_f16(false, af, false, bf1,
                                                      (short)0, acc1, false, false);
      }
#pragma unroll
      for (int nn = 0; nn < 2; ++nn) {
        const v8f& acc = nn ? acc1 : acc0;
        const int n = (nt0 + nn) * 16 + (lane & 15);
        const float bv = b1[n];
#pragma unroll
        for (int r = 0; r < 8; ++r) {
          int m = r + ((lane >> 4) << 3);
          s_hid[m * DHID + n] = (_Float16)fmaxf(acc[r] + bv, 0.f);
        }
      }
    }
    __syncthreads();
    // ---- GEMM2: 32 N-tiles; each wave does 4. K = 256 -> 8 steps ----
    {
      const int m0 = lane & 15;
      const int nt0 = wave * 4;
      v8f acc[4];
#pragma unroll
      for (int nn = 0; nn < 4; ++nn) acc[nn] = (v8f){0.f,0.f,0.f,0.f,0.f,0.f,0.f,0.f};
      for (int ks = 0; ks < DHID / 32; ++ks) {
        v16h af;
#pragma unroll
        for (int e = 0; e < 16; ++e) {
          int k = ks * 32 + ((e >> 3) << 4) + ((lane >> 4) << 3) + (e & 7);
          af[e] = s_hid[m0 * DHID + k];
        }
#pragma unroll
        for (int nn = 0; nn < 4; ++nn) {
          v16h bf = *(const v16h*)(W2p + (((size_t)ks * (2 * DHID / 16) + nt0 + nn) * 32 + lane) * 16);
          acc[nn] = __builtin_amdgcn_wmma_f32_16x16x32_f16(false, af, false, bf,
                                                           (short)0, acc[nn], false, false);
        }
      }
#pragma unroll
      for (int nn = 0; nn < 4; ++nn) {
        const int n = (nt0 + nn) * 16 + (lane & 15);
        const float bv = b2[n];
#pragma unroll
        for (int r = 0; r < 8; ++r) {
          int m = r + ((lane >> 4) << 3);
          s_p[m * (2 * DHID) + n] = acc[nn][r] + bv;
        }
      }
    }
    __syncthreads();
    // ---- logits: 16 j x 8 h ----
    if (t < 16 * HH) {
      int jl = t & 15, h = t >> 4;
      int j = jt * 16 + jl;
      const float* kr = kv + ((size_t)b * JJ + j) * (2 * DHID) + h * DK;
      float a = 0.f;
      for (int d = 0; d < DK; ++d)
        a += s_q[h * DK + d] * (kr[d] + s_p[jl * (2 * DHID) + h * (2 * DK) + d]);
      a *= 0.17677669529663687f;
      if (mask[(size_t)b * JJ + j]) a = -1e9f;
      s_log[jl * HH + h] = a;
    }
    __syncthreads();
    // ---- online softmax state update ----
    if (t < HH) {
      int h = t;
      float mx = s_m[h];
      for (int jl = 0; jl < 16; ++jl) mx = fmaxf(mx, s_log[jl * HH + h]);
      float sc = __expf(s_m[h] - mx);
      float sum = s_s[h] * sc;
      for (int jl = 0; jl < 16; ++jl) {
        float e = __expf(s_log[jl * HH + h] - mx);
        s_w[jl * HH + h] = e;
        sum += e;
      }
      s_m[h] = mx; s_s[h] = sum; s_sc[h] = sc;
    }
    __syncthreads();
    // ---- accumulate V ----
    {
      int h = t >> 5, d = t & 31;
      float a = s_acc[t] * s_sc[h];
      for (int jl = 0; jl < 16; ++jl) {
        int j = jt * 16 + jl;
        a += s_w[jl * HH + h] *
             (kv[((size_t)b * JJ + j) * (2 * DHID) + DHID + h * DK + d] +
              s_p[jl * (2 * DHID) + h * (2 * DK) + DK + d]);
      }
      s_acc[t] = a;
    }
    __syncthreads();
  }
  agg[(size_t)bi * DHID + t] = s_acc[t] / s_s[t >> 5];
}

// ---------------------------------------------------------------------------
// Elementwise add: y = a + c
// ---------------------------------------------------------------------------
__global__ void add_kernel(const float* __restrict__ a, const float* __restrict__ c,
                           float* __restrict__ y, int n) {
  int i = blockIdx.x * blockDim.x + threadIdx.x;
  if (i < n) y[i] = a[i] + c[i];
}

// ---------------------------------------------------------------------------
// Host side
// ---------------------------------------------------------------------------
extern "C" void kernel_launch(void* const* d_in, const int* in_sizes, int n_in,
                              void* d_out, int out_size, void* d_ws, size_t ws_size,
                              hipStream_t stream) {
  (void)in_sizes; (void)n_in; (void)out_size; (void)ws_size;
  const float* dec      = (const float*)d_in[0];
  const float* enc      = (const float*)d_in[1];
  const float* distance = (const float*)d_in[2];
  const float* gdist    = (const float*)d_in[3];
  const float* ldist    = (const float*)d_in[4];
  const int*   lidx     = (const int*)d_in[5];
  const int*   smask    = (const int*)d_in[7];
  const unsigned char* pmask = (const unsigned char*)d_in[8];
  const float* ln_s_g = (const float*)d_in[9],  *ln_s_b = (const float*)d_in[10];
  const float* Wqkv   = (const float*)d_in[11], *Wo_s   = (const float*)d_in[12];
  const float* ln_l_g = (const float*)d_in[13], *ln_l_b = (const float*)d_in[14];
  const float* Wq_l   = (const float*)d_in[15], *Wkv_l  = (const float*)d_in[16];
  const float* pW1_l  = (const float*)d_in[17], *pb1_l  = (const float*)d_in[18];
  const float* pW2_l  = (const float*)d_in[19], *pb2_l  = (const float*)d_in[20];
  const float* Wo_l   = (const float*)d_in[21];
  const float* ln_g_g = (const float*)d_in[22], *ln_g_b = (const float*)d_in[23];
  const float* Wq_g   = (const float*)d_in[24], *Wkv_g  = (const float*)d_in[25];
  const float* pW1_g  = (const float*)d_in[26], *pb1_g  = (const float*)d_in[27];
  const float* pW2_g  = (const float*)d_in[28], *pb2_g  = (const float*)d_in[29];
  const float* Wo_g   = (const float*)d_in[30];
  const float* ln_f_g = (const float*)d_in[31], *ln_f_b = (const float*)d_in[32];
  const float* W1_f   = (const float*)d_in[33], *b1_f   = (const float*)d_in[34];
  const float* W2_f   = (const float*)d_in[35], *b2_f   = (const float*)d_in[36];
  float* out = (float*)d_out;

  // ---- workspace bump allocator ----
  char* wsb = (char*)d_ws;
  size_t off = 0;
  auto alloc = [&](size_t bytes) -> void* {
    void* p = wsb + off;
    off += (bytes + 255) & ~(size_t)255;
    return p;
  };
  auto allocF = [&](size_t n) { return (float*)alloc(n * sizeof(float)); };
  auto allocH = [&](size_t n) { return (_Float16*)alloc(n * sizeof(_Float16)); };

  // packed f16 weights
  _Float16* Wqkv_p = allocH((size_t)DD * 3 * DD);
  _Float16* Wo_s_p = allocH((size_t)DD * DD);
  _Float16* Wq_l_p = allocH((size_t)DD * DHID);
  _Float16* Wkv_l_p = allocH((size_t)DD * 2 * DHID);
  _Float16* pW1_l_p = allocH((size_t)DD * DHID);
  _Float16* pW2_l_p = allocH((size_t)DHID * 2 * DHID);
  _Float16* Wo_l_p = allocH((size_t)(DHID + DD) * DD);
  _Float16* Wq_g_p = allocH((size_t)DD * DHID);
  _Float16* Wkv_g_p = allocH((size_t)DD * 2 * DHID);
  _Float16* pW1_g_p = allocH((size_t)DHID * DHID);
  _Float16* pW2_g_p = allocH((size_t)DHID * 2 * DHID);
  _Float16* Wo_g_p = allocH((size_t)DHID * DD);
  _Float16* W1_f_p = allocH((size_t)DD * DFF);
  _Float16* W2_f_p = allocH((size_t)DFF * DD);

  // activations
  float* xn    = allocF((size_t)NROW * DD);
  float* qkv   = allocF((size_t)NROW * 3 * DD);
  float* sattn = allocF((size_t)NROW * DD);
  float* x1    = allocF((size_t)NROW * DD);
  float* xn_l  = allocF((size_t)NROW * DD);
  float* q_l   = allocF((size_t)NROW * DHID);
  float* kv_l  = allocF((size_t)BB * JJ * 2 * DHID);
  float* hid_l = allocF((size_t)NROW * KK * DHID);
  float* p_l   = allocF((size_t)NROW * KK * 2 * DHID);
  float* ccat  = allocF((size_t)NROW * (DHID + DD));
  float* loco  = allocF((size_t)NROW * DD);
  float* xn_g  = allocF((size_t)NROW * DD);
  float* q_g   = allocF((size_t)NROW * DHID);
  float* kv_g  = allocF((size_t)BB * JJ * 2 * DHID);
  float* agg_g = allocF((size_t)NROW * DHID);
  float* x2    = allocF((size_t)NROW * DD);
  float* x3    = allocF((size_t)NROW * DD);
  float* xf    = allocF((size_t)NROW * DD);
  float* ffh   = allocF((size_t)NROW * DFF);

  auto pack = [&](const float* W, _Float16* P, int Kd, int N) {
    size_t total = (size_t)Kd * N;
    int blocks = (int)((total + 255) / 256);
    if (blocks > 2048) blocks = 2048;
    pack_b_kernel<<<blocks, 256, 0, stream>>>(W, P, Kd, N);
  };
  auto gemm = [&](const float* A, const _Float16* Bp, const float* bias,
                  const float* res, float* C, int M, int Kd, int N, int act) {
    dim3 grid(M / 16, N / 64);
    if (act == 0)      gemm_wmma_kernel<0><<<grid, 128, 0, stream>>>(A, Bp, bias, res, C, M, Kd, N);
    else if (act == 1) gemm_wmma_kernel<1><<<grid, 128, 0, stream>>>(A, Bp, bias, res, C, M, Kd, N);
    else               gemm_wmma_kernel<2><<<grid, 128, 0, stream>>>(A, Bp, bias, res, C, M, Kd, N);
  };

  // ---- pack all weights ----
  pack(Wqkv,  Wqkv_p,  DD, 3 * DD);
  pack(Wo_s,  Wo_s_p,  DD, DD);
  pack(Wq_l,  Wq_l_p,  DD, DHID);
  pack(Wkv_l, Wkv_l_p, DD, 2 * DHID);
  pack(pW1_l, pW1_l_p, DD, DHID);
  pack(pW2_l, pW2_l_p, DHID, 2 * DHID);
  pack(Wo_l,  Wo_l_p,  DHID + DD, DD);
  pack(Wq_g,  Wq_g_p,  DD, DHID);
  pack(Wkv_g, Wkv_g_p, DD, 2 * DHID);
  pack(pW1_g, pW1_g_p, DHID, DHID);
  pack(pW2_g, pW2_g_p, DHID, 2 * DHID);
  pack(Wo_g,  Wo_g_p,  DHID, DD);
  pack(W1_f,  W1_f_p,  DD, DFF);
  pack(W2_f,  W2_f_p,  DFF, DD);

  // ---- self-attention + residual ----
  ln_kernel<<<NROW, 256, 0, stream>>>(dec, ln_s_g, ln_s_b, xn, /*addpe=*/1);
  gemm(xn, Wqkv_p, nullptr, nullptr, qkv, NROW, DD, 3 * DD, 0);
  self_attn_kernel<<<BB * HH, 256, 0, stream>>>(qkv, smask, sattn);
  gemm(sattn, Wo_s_p, nullptr, dec, x1, NROW, DD, DD, 0);

  // ---- local cross-attention branch (no residual; added later) ----
  ln_kernel<<<NROW, 256, 0, stream>>>(x1, ln_l_g, ln_l_b, xn_l, 0);
  gemm(xn_l, Wq_l_p, nullptr, nullptr, q_l, NROW, DD, DHID, 0);
  gemm(enc, Wkv_l_p, nullptr, nullptr, kv_l, BB * JJ, DD, 2 * DHID, 0);
  gemm(ldist, pW1_l_p, pb1_l, nullptr, hid_l, NROW * KK, DD, DHID, 1);     // relu
  gemm(hid_l, pW2_l_p, pb2_l, nullptr, p_l, NROW * KK, DHID, 2 * DHID, 0);
  local_attn_kernel<<<NROW, 256, 0, stream>>>(q_l, kv_l, p_l, lidx, distance, ccat);
  gemm(ccat, Wo_l_p, nullptr, nullptr, loco, NROW, DHID + DD, DD, 0);

  // ---- global cross-attention branch + residual ----
  ln_kernel<<<NROW, 256, 0, stream>>>(x1, ln_g_g, ln_g_b, xn_g, 0);
  gemm(xn_g, Wq_g_p, nullptr, nullptr, q_g, NROW, DD, DHID, 0);
  gemm(enc, Wkv_g_p, nullptr, nullptr, kv_g, BB * JJ, DD, 2 * DHID, 0);
  global_attn_kernel<<<NROW, 256, 0, stream>>>(q_g, kv_g, gdist, pW1_g_p, pb1_g,
                                               pW2_g_p, pb2_g, pmask, agg_g);
  gemm(agg_g, Wo_g_p, nullptr, x1, x2, NROW, DHID, DD, 0);

  // ---- x3 = x2 + local ----
  add_kernel<<<(NROW * DD + 255) / 256, 256, 0, stream>>>(x2, loco, x3, NROW * DD);

  // ---- FFN + residual -> out ----
  ln_kernel<<<NROW, 256, 0, stream>>>(x3, ln_f_g, ln_f_b, xf, 0);
  gemm(xf, W1_f_p, b1_f, nullptr, ffh, NROW, DD, DFF, 2);                  // gelu
  gemm(ffh, W2_f_p, b2_f, x3, out, NROW, DFF, DD, 0);
}
